// BoundaryDecoderLayer_26448408608966
// MI455X (gfx1250) — matrix operand order repaired and checked by reading, not compiled, not run on hardware
//
#include <hip/hip_runtime.h>

typedef __bf16 bf16_t;
typedef __attribute__((ext_vector_type(16))) __bf16 bf16x16;
typedef __attribute__((ext_vector_type(8)))  float  f32x8;
typedef __attribute__((ext_vector_type(2)))  float  f32x2;

#define TT   4096
#define NB   16
#define DD   256
#define NQQ  64
#define MH   8
#define PP   4
#define DHH  32
#define DFFC 2048

#define WMMA_BF16(acc, a, b) \
  acc = __builtin_amdgcn_wmma_f32_16x16x32_bf16(false, a, false, b, (short)0, acc, false, false)

__device__ inline float wave_sum32(float v) {
#pragma unroll
  for (int off = 16; off >= 1; off >>= 1) v += __shfl_xor(v, off, 32);
  return v;
}

// Async global->LDS copy (CDNA5, tracked by ASYNCcnt). lds operand is a byte
// offset into the workgroup's LDS segment (dynamic segment starts at 0).
__device__ inline void async_ld_lds_b32(char* smem_base, void* lds_ptr,
                                        const void* gptr) {
  unsigned loff = (unsigned)((char*)lds_ptr - smem_base);
  asm volatile("global_load_async_to_lds_b32 %0, %1, off"
               :: "v"(loff), "v"(gptr)
               : "memory");
}
__device__ inline void async_wait0() {
  asm volatile("s_wait_asynccnt 0x0" ::: "memory");
}

// Stage KT x NT fragments (each 32x16) of fp32 matrix W (row stride ldb) into
// LDS in WMMA B-operand order, converted to bf16. Coalesced global reads.
// B layout (32x16, 16-bit): lanes 0-15 -> col=lane, K={0..7,16..23};
// lanes 16-31 -> col=lane-16, K={8..15,24..31}.
__device__ inline void load_b_frags(const float* __restrict__ W, int ldb,
                                    int k0, int n0, int KT, int NT,
                                    bf16_t* sm) {
  const int width = NT * 16;
  const int total = KT * 32 * width;
  for (int idx = threadIdx.x; idx < total; idx += blockDim.x) {
    int col  = idx % width;
    int krow = idx / width;
    float val = W[(long)(k0 + krow) * ldb + (n0 + col)];
    int kt = krow >> 5, kin = krow & 31;
    int h  = (kin >> 3) & 1,  j   = (kin & 7) | ((kin >> 4) << 3);
    int nt = col >> 4,  cl  = col & 15;
    int l  = h * 16 + cl;
    sm[(((kt * NT + nt) * 32 + l) << 4) + j] = (bf16_t)val;
  }
}

// A fragment (16x32 bf16) from an fp32 row-major source.
// A layout: lanes 0-15 -> row=lane, K = {0..7,16..23}; lanes 16-31 ->
// row=lane-16, K = {8..15,24..31}; VGPR g holds a consecutive K pair.
__device__ inline bf16x16 load_a_f32(const float* __restrict__ A, long stride,
                                     int k0, int lane, bool stream) {
  int half = lane >> 4, row = lane & 15;
  const float* p = A + (long)row * stride + k0;
  bf16x16 a;
#pragma unroll
  for (int jj = 0; jj < 8; ++jj) {
    int K = 8 * half + ((jj & 4) ? 16 : 0) + 2 * (jj & 3);
    f32x2 f;
    if (stream) f = __builtin_nontemporal_load((const f32x2*)(p + K));
    else        f = *(const f32x2*)(p + K);
    a[2 * jj]     = (bf16_t)f.x;
    a[2 * jj + 1] = (bf16_t)f.y;
  }
  return a;
}

__device__ inline bf16x16 load_a_bf16(const bf16_t* __restrict__ A, long stride,
                                      int k0, int lane) {
  int half = lane >> 4, row = lane & 15;
  const bf16_t* p = A + (long)row * stride + k0;
  bf16x16 a;
#pragma unroll
  for (int jj = 0; jj < 8; ++jj) {
    int K = 8 * half + ((jj & 4) ? 16 : 0) + 2 * (jj & 3);
    a[2 * jj]     = p[K];
    a[2 * jj + 1] = p[K + 1];
  }
  return a;
}

// ---------------------------------------------------------------------------
// Kernel 1: v = features(T,N,512) @ Wv(512,256) + bv, stored head-major as
// v_ws[n][m][t][dh].  Grid 512 blocks (16 n x 32 row-blocks), 8 waves/block.
// Waves tile the 128x256 output as 2x4 groups of 64x64; each wave runs a
// 4x4 WMMA block (16 wmma per k-step from 4 A + 4 B fragments).
// K=512 staged through LDS in 2 chunks of 8 k-tiles (128KB -> 2 blocks/WGP).
// ---------------------------------------------------------------------------
__global__ __launch_bounds__(256) void k_vproj(
    const float* __restrict__ features, const float* __restrict__ Wv,
    const float* __restrict__ bv, float* __restrict__ v_ws) {
  extern __shared__ char smem_raw[];
  bf16_t* bsm = (bf16_t*)smem_raw;                 // 8*16 frags = 128KB
  const int n  = blockIdx.x >> 5;
  const int t0 = (blockIdx.x & 31) * 128;
  const int lane = threadIdx.x & 31;
  const int wave = threadIdx.x >> 5;
  const int rw = wave >> 2;                        // 0..1 : 64-row group
  const int cw = wave & 3;                         // 0..3 : 64-col group
  const int tb = t0 + rw * 64;
  const long astr = (long)NB * 2 * DD;
  const float* Abase = features + ((long)tb * NB + n) * (2 * DD);

  f32x8 z = {0.f, 0.f, 0.f, 0.f, 0.f, 0.f, 0.f, 0.f};
  f32x8 acc[4][4];
#pragma unroll
  for (int x = 0; x < 4; ++x)
#pragma unroll
    for (int y = 0; y < 4; ++y) acc[x][y] = z;

  const bf16x16* frag = (const bf16x16*)bsm;
  for (int chunk = 0; chunk < 2; ++chunk) {
    __syncthreads();
    load_b_frags(Wv, DD, chunk * 256, 0, 8, 16, bsm);
    __syncthreads();
#pragma unroll 2
    for (int kt = 0; kt < 8; ++kt) {
      const int k0 = chunk * 256 + kt * 32;
      bf16x16 a4[4], b4[4];
#pragma unroll
      for (int x = 0; x < 4; ++x)
        a4[x] = load_a_f32(Abase + (long)(x * 16) * astr, astr, k0, lane, true);
#pragma unroll
      for (int y = 0; y < 4; ++y)
        b4[y] = frag[(kt * 16 + cw * 4 + y) * 32 + lane];
#pragma unroll
      for (int x = 0; x < 4; ++x)
#pragma unroll
        for (int y = 0; y < 4; ++y) WMMA_BF16(acc[x][y], a4[x], b4[y]);
    }
  }

  // C layout: lanes 0-15 col=lane rows 0..7; lanes 16-31 col=lane-16 rows 8..15
  const int mrow = (lane >> 4) * 8;
  const int cl   = lane & 15;
#pragma unroll
  for (int x = 0; x < 4; ++x)
#pragma unroll
    for (int y = 0; y < 4; ++y) {
      int c = cw * 64 + y * 16 + cl;
      int head = c >> 5, dh = c & 31;
      float bias = bv[c];
      float* outp =
          v_ws + ((long)(n * MH + head) * TT + tb + x * 16 + mrow) * DHH + dh;
#pragma unroll
      for (int r = 0; r < 8; ++r) outp[r * DHH] = acc[x][y][r] + bias;
    }
}

// ---------------------------------------------------------------------------
// Kernel 2: projections + softmax + bilinear gather + Wo + residual + LN ->
// tgt_ws[(q*16+n)*256 + d].  One block per n.  pro_features staged into LDS
// with CDNA5 async global->LDS loads (ASYNCcnt).
// ---------------------------------------------------------------------------
__global__ __launch_bounds__(256) void k_sample(
    const float* __restrict__ proposal_points, const float* __restrict__ pro_features,
    const float* __restrict__ window_size,
    const float* __restrict__ Wpw, const float* __restrict__ bpw,
    const float* __restrict__ Wpo, const float* __restrict__ bpo,
    const float* __restrict__ Wo,  const float* __restrict__ bo,
    const float* __restrict__ g2,  const float* __restrict__ be2,
    const float* __restrict__ v_ws, float* __restrict__ tgt_ws) {
  extern __shared__ char smem_raw[];
  float* pfs  = (float*)smem_raw;          // 64*256
  float* pts  = pfs + NQQ * DD;            // 64*256
  float* offs = pts + NQQ * DD;            // 64*32
  float* wls  = offs + NQQ * MH * PP;      // 64*32
  const int n   = blockIdx.x;
  const int tid = threadIdx.x;

  // async-stage pf rows for this n into LDS
  for (int i = tid; i < NQQ * DD; i += 256) {
    int q = i >> 8, d = i & 255;
    async_ld_lds_b32(smem_raw, &pfs[i],
                     &pro_features[((long)q * NB + n) * DD + d]);
  }
  async_wait0();
  __syncthreads();

  // off = pf@Wpo + bpo ; wlogits = pf@Wpw + bpw
  for (int i = tid; i < NQQ * 64; i += 256) {
    int q = i >> 6, c = i & 63;
    int cc = c & 31;
    const float* W = (c < 32) ? Wpo : Wpw;
    const float* pr = pfs + q * DD;
    float s = 0.f;
#pragma unroll 4
    for (int k = 0; k < DD; ++k) s += pr[k] * W[k * 32 + cc];
    if (c < 32) offs[q * 32 + cc] = s + bpo[cc];
    else        wls [q * 32 + cc] = s + bpw[cc];
  }
  __syncthreads();

  // softmax over P=4 per (q,m)
  for (int g = tid; g < NQQ * MH; g += 256) {
    float* p = wls + g * 4;
    float m0 = fmaxf(fmaxf(p[0], p[1]), fmaxf(p[2], p[3]));
    float e0 = __expf(p[0] - m0), e1 = __expf(p[1] - m0);
    float e2 = __expf(p[2] - m0), e3 = __expf(p[3] - m0);
    float inv = 1.f / (e0 + e1 + e2 + e3);
    p[0] = e0 * inv; p[1] = e1 * inv; p[2] = e2 * inv; p[3] = e3 * inv;
  }
  __syncthreads();

  // bilinear gather: wave handles (q,m) pairs, lane = dh (coalesced 128B rows)
  const int lane = tid & 31, wave = tid >> 5;
  const float wsn = window_size[n];
  const float scale = wsn * (1.f / TT);
  for (int pair = wave; pair < NQQ * MH; pair += 8) {
    int q = pair >> 3, m = pair & 7;
    float pv = proposal_points[q * NB + n];
    const float* vb = v_ws + (long)(n * MH + m) * TT * DHH;
    float acc = 0.f;
#pragma unroll
    for (int p = 0; p < PP; ++p) {
      float of = offs[q * 32 + m * 4 + p];
      float wg = wls [q * 32 + m * 4 + p];
      float x = pv * scale + of * (1.f / TT);
      x = fminf(fmaxf(x, 0.f), 1.f) * (float)(TT - 1);
      float fl = floorf(x);
      float fr = x - fl;
      int i0 = (int)fl;
      int i1 = min(i0 + 1, TT - 1);
      float g0v = vb[(long)i0 * DHH + lane];
      float g1v = vb[(long)i1 * DHH + lane];
      acc += wg * (g0v + (g1v - g0v) * fr);
    }
    pts[q * DD + lane * MH + m] = acc;   // pt[q][d], d = dh*M + m
  }
  __syncthreads();

  // pt @ Wo + bo  (reuse pfs as output buffer)
  float* pto = pfs;
  for (int i = tid; i < NQQ * DD; i += 256) {
    int q = i >> 8, j = i & 255;
    const float* pr = pts + q * DD;
    float s = 0.f;
#pragma unroll 4
    for (int k = 0; k < DD; ++k) s += pr[k] * Wo[k * DD + j];
    pto[i] = s + bo[j];
  }
  __syncthreads();

  // residual + LayerNorm(g2,be2) -> tgt_ws
  for (int qi = 0; qi < 8; ++qi) {
    int q = wave * 8 + qi;
    long rg = (long)q * NB + n;
    float vals[8];
    float s = 0.f, sq = 0.f;
#pragma unroll
    for (int k = 0; k < 8; ++k) {
      int d = lane + 32 * k;
      float v = pro_features[rg * DD + d] + pto[q * DD + d];
      vals[k] = v; s += v; sq += v * v;
    }
    s = wave_sum32(s); sq = wave_sum32(sq);
    float mean = s * (1.f / DD);
    float var  = sq * (1.f / DD) - mean * mean;
    float rstd = rsqrtf(var + 1e-5f);
#pragma unroll
    for (int k = 0; k < 8; ++k) {
      int d = lane + 32 * k;
      tgt_ws[rg * DD + d] = (vals[k] - mean) * rstd * g2[d] + be2[d];
    }
  }
}

// ---------------------------------------------------------------------------
// Kernel 3: h = relu(tgt @ W1 + b1), stored bf16.  Grid (8 rowblk, 16 colblk).
// Waves tile 128x128 as 2x4 groups of 64x32; 4x2 WMMA block per wave.
// ---------------------------------------------------------------------------
__global__ __launch_bounds__(256) void k_ffn1(
    const float* __restrict__ tgt_ws, const float* __restrict__ W1,
    const float* __restrict__ b1, bf16_t* __restrict__ h_ws) {
  extern __shared__ char smem_raw[];
  bf16_t* bsm = (bf16_t*)smem_raw;                 // 8*8 frags = 64KB
  const int r0 = blockIdx.x * 128;
  const int c0 = blockIdx.y * 128;
  load_b_frags(W1, DFFC, 0, c0, 8, 8, bsm);
  __syncthreads();

  const int lane = threadIdx.x & 31, wave = threadIdx.x >> 5;
  const int rw = wave >> 2;                        // 0..1
  const int cw = wave & 3;                         // 0..3
  const float* Abase = tgt_ws + (long)(r0 + rw * 64) * DD;
  f32x8 z = {0.f, 0.f, 0.f, 0.f, 0.f, 0.f, 0.f, 0.f};
  f32x8 acc[4][2];
#pragma unroll
  for (int x = 0; x < 4; ++x) { acc[x][0] = z; acc[x][1] = z; }
  const bf16x16* frag = (const bf16x16*)bsm;
#pragma unroll 2
  for (int kt = 0; kt < 8; ++kt) {
    bf16x16 a4[4], b2[2];
#pragma unroll
    for (int x = 0; x < 4; ++x)
      a4[x] = load_a_f32(Abase + (long)(x * 16) * DD, DD, kt * 32, lane, false);
#pragma unroll
    for (int y = 0; y < 2; ++y)
      b2[y] = frag[(kt * 8 + cw * 2 + y) * 32 + lane];
#pragma unroll
    for (int x = 0; x < 4; ++x)
#pragma unroll
      for (int y = 0; y < 2; ++y) WMMA_BF16(acc[x][y], a4[x], b2[y]);
  }

  const int mrow = (lane >> 4) * 8, cl = lane & 15;
#pragma unroll
  for (int x = 0; x < 4; ++x)
#pragma unroll
    for (int y = 0; y < 2; ++y) {
      int c = c0 + cw * 32 + y * 16 + cl;
      float bias = b1[c];
      bf16_t* outp = h_ws + (long)(r0 + rw * 64 + x * 16 + mrow) * DFFC + c;
#pragma unroll
      for (int r = 0; r < 8; ++r) {
        float v = acc[x][y][r] + bias;
        outp[(long)r * DFFC] = (bf16_t)fmaxf(v, 0.f);
      }
    }
}

// ---------------------------------------------------------------------------
// Kernel 4: out = LayerNorm(tgt + h @ W2 + b2, g3, be3).  K=2048 staged in 8
// LDS chunks of 8 k-tiles (128KB); waves tile 128x256 as 2x4 groups of 64x64
// with 4x4 WMMA blocks.  Epilogue reuses LDS for the 128x256 tile + LN.
// ---------------------------------------------------------------------------
__global__ __launch_bounds__(256) void k_ffn2(
    const bf16_t* __restrict__ h_ws, const float* __restrict__ W2,
    const float* __restrict__ b2, const float* __restrict__ tgt_ws,
    const float* __restrict__ g3, const float* __restrict__ be3,
    float* __restrict__ out) {
  extern __shared__ char smem_raw[];
  bf16_t* bsm = (bf16_t*)smem_raw;                 // 8*16 frags = 128KB
  const int r0 = blockIdx.x * 128;
  const int lane = threadIdx.x & 31, wave = threadIdx.x >> 5;
  const int rw = wave >> 2;                        // 0..1
  const int cw = wave & 3;                         // 0..3
  const bf16_t* Abase = h_ws + (long)(r0 + rw * 64) * DFFC;

  f32x8 z = {0.f, 0.f, 0.f, 0.f, 0.f, 0.f, 0.f, 0.f};
  f32x8 acc[4][4];
#pragma unroll
  for (int x = 0; x < 4; ++x)
#pragma unroll
    for (int y = 0; y < 4; ++y) acc[x][y] = z;
  const bf16x16* frag = (const bf16x16*)bsm;

  for (int chunk = 0; chunk < 8; ++chunk) {
    __syncthreads();
    load_b_frags(W2, DD, chunk * 256, 0, 8, 16, bsm);
    __syncthreads();
#pragma unroll 2
    for (int kt = 0; kt < 8; ++kt) {
      const int k0 = chunk * 256 + kt * 32;
      bf16x16 a4[4], b4[4];
#pragma unroll
      for (int x = 0; x < 4; ++x)
        a4[x] = load_a_bf16(Abase + (long)(x * 16) * DFFC, DFFC, k0, lane);
#pragma unroll
      for (int y = 0; y < 4; ++y)
        b4[y] = frag[(kt * 16 + cw * 4 + y) * 32 + lane];
#pragma unroll
      for (int x = 0; x < 4; ++x)
#pragma unroll
        for (int y = 0; y < 4; ++y) WMMA_BF16(acc[x][y], a4[x], b4[y]);
    }
  }
  __syncthreads();

  float* fs = (float*)smem_raw;                    // 128 x 256 tile (128KB)
  const int mrow = (lane >> 4) * 8, cl = lane & 15;
#pragma unroll
  for (int x = 0; x < 4; ++x)
#pragma unroll
    for (int y = 0; y < 4; ++y) {
      int c = cw * 64 + y * 16 + cl;
      float bias = b2[c];
#pragma unroll
      for (int r = 0; r < 8; ++r)
        fs[(rw * 64 + x * 16 + mrow + r) * DD + c] = acc[x][y][r] + bias;
    }
  __syncthreads();

  for (int ri = 0; ri < 16; ++ri) {
    int rl = wave * 16 + ri;
    long rg = r0 + rl;
    float vals[8];
    float s = 0.f, sq = 0.f;
#pragma unroll
    for (int k = 0; k < 8; ++k) {
      int d = lane + 32 * k;
      float v = fs[rl * DD + d] + tgt_ws[rg * DD + d];
      vals[k] = v; s += v; sq += v * v;
    }
    s = wave_sum32(s); sq = wave_sum32(sq);
    float mean = s * (1.f / DD);
    float var  = sq * (1.f / DD) - mean * mean;
    float rstd = rsqrtf(var + 1e-5f);
#pragma unroll
    for (int k = 0; k < 8; ++k) {
      int d = lane + 32 * k;
      out[rg * DD + d] = (vals[k] - mean) * rstd * g3[d] + be3[d];
    }
  }
}

// ---------------------------------------------------------------------------
extern "C" void kernel_launch(void* const* d_in, const int* in_sizes, int n_in,
                              void* d_out, int out_size, void* d_ws, size_t ws_size,
                              hipStream_t stream) {
  const float* features        = (const float*)d_in[0];
  const float* proposal_points = (const float*)d_in[1];
  const float* pro_features    = (const float*)d_in[2];
  const float* window_size     = (const float*)d_in[3];
  const float* Wv  = (const float*)d_in[4];
  const float* bv  = (const float*)d_in[5];
  const float* Wpw = (const float*)d_in[6];
  const float* bpw = (const float*)d_in[7];
  const float* Wpo = (const float*)d_in[8];
  const float* bpo = (const float*)d_in[9];
  const float* Wo  = (const float*)d_in[10];
  const float* bo  = (const float*)d_in[11];
  const float* W1  = (const float*)d_in[12];
  const float* b1  = (const float*)d_in[13];
  const float* W2  = (const float*)d_in[14];
  const float* b2  = (const float*)d_in[15];
  const float* g2  = (const float*)d_in[16];
  const float* be2 = (const float*)d_in[17];
  const float* g3  = (const float*)d_in[18];
  const float* be3 = (const float*)d_in[19];

  float*  v_ws   = (float*)d_ws;                                   // 64 MB
  float*  tgt_ws = (float*)((char*)d_ws + (size_t)67108864);       // 1 MB
  bf16_t* h_ws   = (bf16_t*)((char*)d_ws + (size_t)68157440);      // 4 MB

  hipFuncSetAttribute((const void*)k_vproj,  hipFuncAttributeMaxDynamicSharedMemorySize, 131072);
  hipFuncSetAttribute((const void*)k_sample, hipFuncAttributeMaxDynamicSharedMemorySize, 147456);
  hipFuncSetAttribute((const void*)k_ffn1,   hipFuncAttributeMaxDynamicSharedMemorySize, 65536);
  hipFuncSetAttribute((const void*)k_ffn2,   hipFuncAttributeMaxDynamicSharedMemorySize, 131072);

  k_vproj<<<dim3(512), dim3(256), 131072, stream>>>(features, Wv, bv, v_ws);
  k_sample<<<dim3(16), dim3(256), 147456, stream>>>(proposal_points, pro_features,
      window_size, Wpw, bpw, Wpo, bpo, Wo, bo, g2, be2, v_ws, tgt_ws);
  k_ffn1<<<dim3(8, 16), dim3(256), 65536, stream>>>(tgt_ws, W1, b1, h_ws);
  k_ffn2<<<dim3(8), dim3(256), 131072, stream>>>(h_ws, W2, b2, tgt_ws, g3, be3,
                                                 (float*)d_out);
}
